// LinkGNN_16853451670012
// MI455X (gfx1250) — matrix operand order repaired and checked by reading, not compiled, not run on hardware
//
#include <hip/hip_runtime.h>
#include <hip/hip_bf16.h>

typedef __attribute__((ext_vector_type(16))) _Float16 v16h;
typedef __attribute__((ext_vector_type(8)))  _Float16 v8h;
typedef __attribute__((ext_vector_type(8)))  float    v8f;

// ---------------------------------------------------------------- utilities

__global__ void gcn_init_deg(unsigned* __restrict__ deg, int N) {
  int i = blockIdx.x * blockDim.x + threadIdx.x;
  if (i < N) deg[i] = 1u;  // self-loop contributes 1 to every degree
}

__global__ void gcn_zero_f32(float* __restrict__ p, long long n) {
  long long i = (long long)blockIdx.x * blockDim.x + threadIdx.x;
  long long stride = (long long)gridDim.x * blockDim.x;
  for (; i < n; i += stride) p[i] = 0.0f;
}

__global__ void gcn_accum_deg(const int* __restrict__ col,
                              unsigned* __restrict__ deg, long long E) {
  long long i = (long long)blockIdx.x * blockDim.x + threadIdx.x;
  if (i < E) atomicAdd(&deg[col[i]], 1u);
}

__global__ void gcn_dinv(float* __restrict__ dinv_deg, int N) {
  int i = blockIdx.x * blockDim.x + threadIdx.x;
  if (i < N) {
    unsigned d = ((const unsigned*)dinv_deg)[i];
    dinv_deg[i] = rsqrtf((float)d);   // d >= 1 always (self-loop)
  }
}

__global__ void gcn_cvt_f16(const float* __restrict__ in,
                            _Float16* __restrict__ out, long long n) {
  long long i = (long long)blockIdx.x * blockDim.x + threadIdx.x;
  long long stride = (long long)gridDim.x * blockDim.x;
  for (; i < n; i += stride) out[i] = (_Float16)in[i];
}

// -------------------------------------------------- B-matrix pre-pack (f32->f16)
// Packs row-major B[K][Nout] (f32) into WMMA fragment order:
//   Bp[nt][ks][lane][j] (16 halfs per lane = one v16h fragment)
// Layout used (A matches): lane<16 -> column n=nt*16+lane, K = ks*32 + j
//                          lane>=16 -> same column set, K = ks*32 + 16 + j
__global__ void gcn_pack_b(const float* __restrict__ B,
                           _Float16* __restrict__ Bp, int K, int Nout) {
  int ksteps = K >> 5;
  int ntiles = Nout >> 4;
  int total  = ntiles * ksteps * 32;
  int idx = blockIdx.x * blockDim.x + threadIdx.x;
  if (idx >= total) return;
  int lane = idx & 31;
  int ks   = (idx >> 5) % ksteps;
  int nt   = (idx >> 5) / ksteps;
  int hi = lane >> 4, lo16 = lane & 15;
  int n  = nt * 16 + lo16;
  int k0 = ks * 32 + hi * 16;
  v16h b;
#pragma unroll
  for (int j = 0; j < 16; ++j) b[j] = (_Float16)B[(long long)(k0 + j) * Nout + n];
  *(v16h*)(Bp + (long long)idx * 16) = b;
}

// ---------------------------------------------------------------- WMMA GEMM
// D[M,Nout] = A[M,K] (f16, row-major) * B (pre-packed fragments), f32 accum.
// Each wave computes a 16x(16*NT) output slab; 8 waves (8 M-tiles) per block.
template <int NT>
__global__ void gcn_gemm_wmma(const _Float16* __restrict__ A,
                              const _Float16* __restrict__ Bp,
                              float* __restrict__ D,
                              int M, int K, int Nout) {
  const int lane = threadIdx.x & 31;
  const int warp = threadIdx.x >> 5;
  const int hi   = lane >> 4;        // 0: lanes 0-15, 1: lanes 16-31
  const int lo16 = lane & 15;
  const int m0 = (blockIdx.x * (blockDim.x >> 5) + warp) * 16;
  const int ntBase = blockIdx.y * NT;              // in 16-col tile units
  if (m0 >= M) return;

  const int ksteps = K >> 5;
  const long long fragPerTile = (long long)ksteps * 32;  // fragments per n-tile

  int mrow = m0 + lo16;
  int mload = mrow < M ? mrow : (M - 1);   // clamp loads, keep EXEC full

  v8f acc[NT];
#pragma unroll
  for (int t = 0; t < NT; ++t) acc[t] = (v8f){};

  for (int ks = 0; ks < ksteps; ++ks) {
    // A fragment: row mload; lanes 0-15 take K = 32ks+{0..15},
    //             lanes 16-31 take K = 32ks+{16..31}
    const _Float16* ap = A + (long long)mload * K + ks * 32 + hi * 16;
    v8h alo = *(const v8h*)(ap);
    v8h ahi = *(const v8h*)(ap + 8);
    v16h a;
#pragma unroll
    for (int j = 0; j < 8; ++j) { a[j] = alo[j]; a[j + 8] = ahi[j]; }

    const _Float16* bp =
        Bp + (((long long)ntBase * ksteps + ks) * 32 + lane) * 16;
#pragma unroll
    for (int t = 0; t < NT; ++t) {
      v16h b = *(const v16h*)(bp + t * fragPerTile * 16);
      acc[t] = __builtin_amdgcn_wmma_f32_16x16x32_f16(
          false, a, false, b, (short)0, acc[t], false, false);
    }
  }

#pragma unroll
  for (int t = 0; t < NT; ++t) {
    int n0 = (ntBase + t) * 16;
#pragma unroll
    for (int r = 0; r < 8; ++r) {
      int mr = m0 + r + hi * 8;
      if (mr < M) D[(long long)mr * Nout + n0 + lo16] = acc[t][r];
    }
  }
}

// ------------------------------------------------------- edge gather/scatter
// One wave32 per edge: agg[dst] += h[src] * dinv[src]*dinv[dst]
__global__ void gcn_scatter(const float* __restrict__ h,
                            const float* __restrict__ dinv,
                            const int* __restrict__ src,
                            const int* __restrict__ dst,
                            float* __restrict__ agg,
                            long long E, int F) {
  long long wid = ((long long)blockIdx.x * blockDim.x + threadIdx.x) >> 5;
  int lane = threadIdx.x & 31;
  if (wid >= E) return;
  long long r = (long long)src[wid];
  long long c = (long long)dst[wid];
  float norm = dinv[r] * dinv[c];
  if (F == 128) {
    float4 v = ((const float4*)(h + r * 128))[lane];
    float* o = agg + c * 128 + (long long)lane * 4;
    unsafeAtomicAdd(o + 0, v.x * norm);
    unsafeAtomicAdd(o + 1, v.y * norm);
    unsafeAtomicAdd(o + 2, v.z * norm);
    unsafeAtomicAdd(o + 3, v.w * norm);
  } else if (F == 64) {
    float2 v = ((const float2*)(h + r * 64))[lane];
    float* o = agg + c * 64 + (long long)lane * 2;
    unsafeAtomicAdd(o + 0, v.x * norm);
    unsafeAtomicAdd(o + 1, v.y * norm);
  } else {
    for (int f = lane; f < F; f += 32)
      unsafeAtomicAdd(&agg[c * F + f], h[r * F + f] * norm);
  }
}

// -------------------------------------------------------------- finalizers
// out16 = f16( ELU( agg + h*dinv^2 (self-loop) + bias ) )
__global__ void gcn_finalize_elu(const float* __restrict__ agg,
                                 const float* __restrict__ h,
                                 const float* __restrict__ dinv,
                                 const float* __restrict__ bias,
                                 _Float16* __restrict__ out16,
                                 long long total, int F) {
  long long i = (long long)blockIdx.x * blockDim.x + threadIdx.x;
  long long stride = (long long)gridDim.x * blockDim.x;
  for (; i < total; i += stride) {
    long long node = i / F;
    int f = (int)(i - node * F);
    float s = dinv[node];
    float v = agg[i] + h[i] * s * s + bias[f];
    v = v > 0.0f ? v : expm1f(v);
    out16[i] = (_Float16)v;
  }
}

// out = out(agg) + h*dinv^2 + bias   (final layer, fp32 result in d_out)
__global__ void gcn_finalize_out(float* __restrict__ out,
                                 const float* __restrict__ h,
                                 const float* __restrict__ dinv,
                                 const float* __restrict__ bias,
                                 long long total, int F) {
  long long i = (long long)blockIdx.x * blockDim.x + threadIdx.x;
  long long stride = (long long)gridDim.x * blockDim.x;
  for (; i < total; i += stride) {
    long long node = i / F;
    int f = (int)(i - node * F);
    float s = dinv[node];
    out[i] = out[i] + h[i] * s * s + bias[f];
  }
}

// ---------------------------------------------------------------- launcher

extern "C" void kernel_launch(void* const* d_in, const int* in_sizes, int n_in,
                              void* d_out, int out_size, void* d_ws, size_t ws_size,
                              hipStream_t stream) {
  const float* x  = (const float*)d_in[0];
  const int*   ei = (const int*)d_in[1];     // [2,E] int32 (JAX default x64 off)
  const float* W1 = (const float*)d_in[2];
  const float* b1 = (const float*)d_in[3];
  const float* W2 = (const float*)d_in[4];
  const float* b2 = (const float*)d_in[5];

  const int H = in_sizes[3];                 // 128
  const int C = in_sizes[5];                 // 64
  const int F = in_sizes[2] / H;             // 128
  const int N = in_sizes[0] / F;             // 50000
  const long long E = (long long)in_sizes[1] / 2;
  const int* e_src = ei;                     // edge_index[0] : source j
  const int* e_dst = ei + E;                 // edge_index[1] : target i

  // ---- workspace carving (256B aligned) ----
  char* ws = (char*)d_ws;
  size_t off = 0;
  auto carve = [&](size_t bytes) -> char* {
    char* p = ws + off;
    off = (off + bytes + 255) & ~(size_t)255;
    return p;
  };
  float*    dinv = (float*)   carve((size_t)N * 4);
  _Float16* xh   = (_Float16*)carve((size_t)N * F * 2);
  _Float16* w1p  = (_Float16*)carve((size_t)F * H * 2);   // packed fragments
  _Float16* w2p  = (_Float16*)carve((size_t)H * C * 2);   // packed fragments
  float*    h1   = (float*)   carve((size_t)N * H * 4);
  float*    agg1 = (float*)   carve((size_t)N * H * 4);
  float*    h2   = (float*)   carve((size_t)N * C * 4);
  _Float16* h1h  = (H <= F) ? xh : (_Float16*)carve((size_t)N * H * 2);
  (void)ws_size;

  float* out = (float*)d_out;

  auto cdiv = [](long long a, long long b) { return (a + b - 1) / b; };

  // 1) degrees (self-loop seeded), normalization
  gcn_init_deg<<<cdiv(N, 256), 256, 0, stream>>>((unsigned*)dinv, N);
  gcn_accum_deg<<<cdiv(E, 256), 256, 0, stream>>>(e_dst, (unsigned*)dinv, E);
  gcn_dinv<<<cdiv(N, 256), 256, 0, stream>>>(dinv, N);

  // 2) zero accumulators (must happen every call)
  gcn_zero_f32<<<2048, 256, 0, stream>>>(agg1, (long long)N * H);
  gcn_zero_f32<<<2048, 256, 0, stream>>>(out, (long long)N * C);

  // 3) operand prep: x -> f16, W1/W2 -> packed f16 WMMA fragments
  gcn_cvt_f16<<<2048, 256, 0, stream>>>(x, xh, (long long)N * F);
  gcn_pack_b<<<cdiv((long long)(H / 16) * (F / 32) * 32, 256), 256, 0, stream>>>(
      W1, w1p, F, H);
  gcn_pack_b<<<cdiv((long long)(C / 16) * (H / 32) * 32, 256), 256, 0, stream>>>(
      W2, w2p, H, C);

  // 4) layer 1: h1 = x @ W1  (WMMA, 16x64 per wave)
  {
    dim3 grid((unsigned)cdiv(cdiv(N, 16), 8), (unsigned)(H / 64));
    gcn_gemm_wmma<4><<<grid, 256, 0, stream>>>(xh, w1p, h1, N, F, H);
  }
  // 5) edge scatter into agg1
  gcn_scatter<<<cdiv(E, 8), 256, 0, stream>>>(h1, dinv, e_src, e_dst, agg1, E, H);
  // 6) + self-loop + bias, ELU, convert to f16
  gcn_finalize_elu<<<2048, 256, 0, stream>>>(agg1, h1, dinv, b1, h1h,
                                             (long long)N * H, H);

  // 7) layer 2: h2 = elu(h1) @ W2  (WMMA, 16x64 per wave)
  {
    dim3 grid((unsigned)cdiv(cdiv(N, 16), 8), (unsigned)(C / 64));
    gcn_gemm_wmma<4><<<grid, 256, 0, stream>>>(h1h, w2p, h2, N, H, C);
  }
  // 8) edge scatter into d_out
  gcn_scatter<<<cdiv(E, 8), 256, 0, stream>>>(h2, dinv, e_src, e_dst, out, E, C);
  // 9) + self-loop + bias
  gcn_finalize_out<<<2048, 256, 0, stream>>>(out, h2, dinv, b2,
                                             (long long)N * C, C);
}
